// DynamicMemoryBank_76871324664388
// MI455X (gfx1250) — matrix operand order repaired and checked by reading, not compiled, not run on hardware
//
#include <hip/hip_runtime.h>

// ---------------- problem constants ----------------
static constexpr int kB = 64;
static constexpr int kT = 1024;
static constexpr int kC = 256;   // C
static constexpr int kM = 1024;  // M == T
static constexpr int kF = 1024;  // 4C
static constexpr int kK1 = 512;  // 2C

// ---------------- WMMA types ----------------
typedef __attribute__((ext_vector_type(16))) __bf16 v16bf;
typedef __attribute__((ext_vector_type(8)))  float  v8f;
typedef __attribute__((ext_vector_type(8)))  unsigned int v8u;

__device__ __forceinline__ unsigned short f2bf(float f) {
  unsigned u = __builtin_bit_cast(unsigned, f);
  u += 0x7FFFu + ((u >> 16) & 1u);          // round-to-nearest-even
  return (unsigned short)(u >> 16);
}
__device__ __forceinline__ unsigned pack2(float a, float b) {
  return (unsigned)f2bf(a) | ((unsigned)f2bf(b) << 16);
}
__device__ __forceinline__ float bf2f(unsigned short h) {
  unsigned u = ((unsigned)h) << 16;
  return __builtin_bit_cast(float, u);
}

// A operand: 16x32 bf16, row-major source of K-pairs (dword = 2 bf16).
// Layout: lanes0-15 row M, VGPR0-7 K=0..15 (hi half: K=16..31), matching load_B_pairs.
__device__ __forceinline__ v16bf load_A_pairs(const unsigned* p, int ldp, int k0p, int lane) {
  int row = lane & 15, hi = lane >> 4;
  const unsigned* q = p + row * ldp + k0p + hi * 8;
  v8u a;
  a[0] = q[0];  a[1] = q[1];  a[2] = q[2];  a[3] = q[3];
  a[4] = q[4];  a[5] = q[5];  a[6] = q[6];  a[7] = q[7];
  return __builtin_bit_cast(v16bf, a);
}

// B operand: 32x16 bf16 from pre-transposed matrix Bt[N][K] (dword = 2 bf16 along K).
// lanes0-15: cols, K=0..15; lanes16-31: cols, K=16..31 -> 8 contiguous dwords per lane.
__device__ __forceinline__ v16bf load_B_pairs(const unsigned* bt, int ldp, int n0, int k0p, int lane) {
  int n = n0 + (lane & 15), kh = lane >> 4;
  const unsigned* q = bt + (size_t)n * ldp + k0p + kh * 8;
  v8u b;
#pragma unroll
  for (int j = 0; j < 8; ++j) b[j] = q[j];
  return __builtin_bit_cast(v16bf, b);
}

__device__ __forceinline__ v8f wmma_bf16(v16bf a, v16bf b, v8f c) {
  return __builtin_amdgcn_wmma_f32_16x16x32_bf16(false, a, false, b, (short)0, c, false, false);
}

// ---------------- kernel 0: bf16 operand prep ----------------
__global__ __launch_bounds__(256) void dmb_prep_kernel(
    const float* __restrict__ mem, const float* __restrict__ w1, const float* __restrict__ w2,
    unsigned short* __restrict__ memK, unsigned short* __restrict__ memT,
    unsigned short* __restrict__ w1T, unsigned short* __restrict__ w2T) {
  int idx = blockIdx.x * 256 + threadIdx.x;
  if (idx < kM * kC) {                              // memory [M,C] -> memK [M,C], memT [C,M]
    int m = idx >> 8, c = idx & 255;
    unsigned short v = f2bf(mem[idx]);
    memK[idx] = v;
    memT[c * kM + m] = v;
  } else if (idx < kM * kC + kK1 * kF) {            // w1 [512,1024] -> w1T [1024,512]
    int i = idx - kM * kC;
    int k = i >> 10, f = i & 1023;
    w1T[f * kK1 + k] = f2bf(w1[i]);
  } else {                                          // w2 [1024,256] -> w2T [256,1024]
    int i = idx - (kM * kC + kK1 * kF);
    int f = i >> 8, c = i & 255;
    w2T[c * kF + f] = f2bf(w2[i]);
  }
}

// ---------------- kernel 1: x mean over T ----------------
__global__ __launch_bounds__(256) void dmb_xmean_kernel(const float* __restrict__ x,
                                                        float* __restrict__ xmean) {
  int idx = blockIdx.x * 256 + threadIdx.x;   // B*C = 16384
  int b = idx >> 8, c = idx & 255;
  const float* xp = x + (size_t)b * kT * kC + c;
  float s = 0.f;
  for (int t = 0; t < kT; ++t) s += xp[t * kC];
  xmean[idx] = s * (1.f / kT);
}

// ---------------- kernel 2: attention (energy -> softmax -> retrieved) ----------------
__global__ __launch_bounds__(256) void dmb_attn_kernel(
    const float* __restrict__ x, const unsigned* __restrict__ memK,
    const unsigned* __restrict__ memT, float* __restrict__ out /* cat [B,T,2C] */) {
  __shared__ __align__(16) unsigned xA[16 * 128];        // 16 rows x 256 bf16 (pairs)   8KB
  __shared__ __align__(16) float    E[16 * 1024];        // energy tile                 64KB
  __shared__ __align__(16) unsigned short P[16 * 1024];  // probabilities bf16          32KB
  __shared__ float red[16 * 16];
  __shared__ float rowMax[16];
  __shared__ float rowSumInv[16];

  int tile = blockIdx.x;                 // B*T/16 = 4096
  int b = tile >> 6;
  int t0 = (tile & 63) << 4;
  int tid = threadIdx.x;
  int lane = tid & 31, w = tid >> 5;
  int hi = lane >> 4, l15 = lane & 15;
  size_t rowbase = (size_t)b * kT + t0;

  // stage x tile -> cat passthrough + bf16 LDS
  {
    int base = tid * 16;
    int r = base >> 8, c0 = base & 255;
    const float* xp = x + (rowbase + r) * kC + c0;
    float* op = out + (rowbase + r) * (2 * kC) + c0;
#pragma unroll
    for (int i = 0; i < 16; i += 2) {
      float a0 = xp[i], a1 = xp[i + 1];
      op[i] = a0; op[i + 1] = a1;
      xA[r * 128 + ((c0 + i) >> 1)] = pack2(a0, a1);
    }
  }
  __syncthreads();

  // energy: wave w owns m in [w*128, w*128+128): 8 n-tiles, K=256 -> 8 k-steps
  v8f zero = {};
  v8f acc[8];
#pragma unroll
  for (int i = 0; i < 8; ++i) acc[i] = zero;
  int mB = w << 7;
#pragma unroll
  for (int k = 0; k < 8; ++k) {
    v16bf A = load_A_pairs(xA, 128, k * 16, lane);
#pragma unroll
    for (int nt = 0; nt < 8; ++nt) {
      v16bf Bv = load_B_pairs(memK, 128, mB + nt * 16, k * 16, lane);
      acc[nt] = wmma_bf16(A, Bv, acc[nt]);
    }
  }
#pragma unroll
  for (int nt = 0; nt < 8; ++nt) {
    int m0 = mB + nt * 16 + l15;
#pragma unroll
    for (int r = 0; r < 8; ++r) E[(r + 8 * hi) * 1024 + m0] = acc[nt][r];
  }
  __syncthreads();

  // softmax over M (two-pass, per-row, 16 threads per row)
  int srow = tid >> 4, seg = tid & 15;
  {
    const float* e = E + srow * 1024 + seg * 64;
    float mx = -3.4e38f;
    for (int i = 0; i < 64; ++i) mx = fmaxf(mx, e[i]);
    red[srow * 16 + seg] = mx;
  }
  __syncthreads();
  if (seg == 0) {
    float mx = red[srow * 16];
    for (int i = 1; i < 16; ++i) mx = fmaxf(mx, red[srow * 16 + i]);
    rowMax[srow] = mx;
  }
  __syncthreads();
  {
    float mx = rowMax[srow];
    const float* e = E + srow * 1024 + seg * 64;
    unsigned short* pp = P + srow * 1024 + seg * 64;
    float s = 0.f;
    for (int i = 0; i < 64; ++i) {
      float pv = __expf(e[i] - mx);
      s += pv;
      pp[i] = f2bf(pv);
    }
    red[srow * 16 + seg] = s;
  }
  __syncthreads();
  if (seg == 0) {
    float s = 0.f;
    for (int i = 0; i < 16; ++i) s += red[srow * 16 + i];
    rowSumInv[srow] = 1.f / s;
  }
  __syncthreads();

  // retrieved: wave w owns c in [w*32, w*32+32), K = M = 1024 -> 32 k-steps
  v8f racc0 = zero, racc1 = zero;
  int cB = w << 5;
  const unsigned* Pu = (const unsigned*)P;
  for (int kb = 0; kb < 32; ++kb) {
    v16bf A = load_A_pairs(Pu, 512, kb * 16, lane);
    v16bf B0 = load_B_pairs(memT, 512, cB, kb * 16, lane);
    racc0 = wmma_bf16(A, B0, racc0);
    v16bf B1 = load_B_pairs(memT, 512, cB + 16, kb * 16, lane);
    racc1 = wmma_bf16(A, B1, racc1);
  }
#pragma unroll
  for (int r = 0; r < 8; ++r) {
    int row = r + 8 * hi;
    float inv = rowSumInv[row];
    out[(rowbase + row) * (2 * kC) + kC + cB + l15] = racc0[r] * inv;
    out[(rowbase + row) * (2 * kC) + kC + cB + 16 + l15] = racc1[r] * inv;
  }
}

// ---------------- kernel 3: gate MLP ----------------
__global__ __launch_bounds__(256) void dmb_mlp_kernel(
    const float* __restrict__ cat, const unsigned* __restrict__ w1T,
    const unsigned* __restrict__ w2T, const float* __restrict__ b1,
    const float* __restrict__ b2, unsigned short* __restrict__ gate) {
  __shared__ __align__(16) unsigned catA[16 * 256];       // 16 x 512 bf16 (pairs) 16KB
  __shared__ __align__(16) unsigned short hA[16 * 1024];  // 16 x 1024 bf16        32KB

  int tile = blockIdx.x;
  int b = tile >> 6;
  int t0 = (tile & 63) << 4;
  int tid = threadIdx.x;
  int lane = tid & 31, w = tid >> 5;
  int hi = lane >> 4, l15 = lane & 15;
  size_t rowbase = (size_t)b * kT + t0;

  // stage cat tile -> bf16 LDS
  {
    int base = tid * 32;
    int r = base >> 9, k0 = base & 511;
    const float* cp = cat + (rowbase + r) * kK1 + k0;
#pragma unroll
    for (int i = 0; i < 32; i += 2)
      catA[r * 256 + ((k0 + i) >> 1)] = pack2(cp[i], cp[i + 1]);
  }
  __syncthreads();

  // h = silu(cat @ w1 + b1): wave w owns f in [w*128, ...), K = 512 -> 16 k-steps
  v8f zero = {};
  v8f acc[8];
#pragma unroll
  for (int i = 0; i < 8; ++i) acc[i] = zero;
  int fB = w << 7;
  for (int k = 0; k < 16; ++k) {
    v16bf A = load_A_pairs(catA, 256, k * 16, lane);
#pragma unroll
    for (int nt = 0; nt < 8; ++nt) {
      v16bf Bv = load_B_pairs(w1T, 256, fB + nt * 16, k * 16, lane);
      acc[nt] = wmma_bf16(A, Bv, acc[nt]);
    }
  }
#pragma unroll
  for (int nt = 0; nt < 8; ++nt) {
    int f = fB + nt * 16 + l15;
    float bias = b1[f];
#pragma unroll
    for (int r = 0; r < 8; ++r) {
      int row = r + 8 * hi;
      float v = acc[nt][r] + bias;
      float s = v / (1.f + __expf(-v));  // SiLU
      hA[row * 1024 + f] = f2bf(s);
    }
  }
  __syncthreads();

  // gate = sigmoid(h @ w2 + b2): wave w owns c in [w*32, ...), K = 1024 -> 32 k-steps
  v8f g0 = zero, g1 = zero;
  int cB = w << 5;
  const unsigned* Hu = (const unsigned*)hA;
  for (int kb = 0; kb < 32; ++kb) {
    v16bf A = load_A_pairs(Hu, 512, kb * 16, lane);
    v16bf B0 = load_B_pairs(w2T, 512, cB, kb * 16, lane);
    g0 = wmma_bf16(A, B0, g0);
    v16bf B1 = load_B_pairs(w2T, 512, cB + 16, kb * 16, lane);
    g1 = wmma_bf16(A, B1, g1);
  }
#pragma unroll
  for (int nt = 0; nt < 2; ++nt) {
    v8f gv = nt ? g1 : g0;
    int c = cB + nt * 16 + l15;
    float bias = b2[c];
#pragma unroll
    for (int r = 0; r < 8; ++r) {
      int row = r + 8 * hi;
      float v = gv[r] + bias;
      float g = 1.f / (1.f + __expf(-v));
      gate[(rowbase + row) * kC + c] = f2bf(g);
    }
  }
}

// ---------------- kernel 4: memory blend + batch mean (deterministic) ----------------
__global__ __launch_bounds__(256) void dmb_update_kernel(
    const float* __restrict__ memory, const unsigned short* __restrict__ gate,
    const float* __restrict__ xmean, float* __restrict__ outNew) {
  int idx = blockIdx.x * 256 + threadIdx.x;  // M*C = 262144
  int c = idx & 255;
  float mv = memory[idx];
  float s = 0.f;
  for (int b = 0; b < kB; ++b) {
    float g = bf2f(gate[(size_t)b * (kM * kC) + idx]);
    s += mv * (1.f - g) + xmean[b * kC + c] * g;
  }
  outNew[idx] = s * (1.f / kB);
}

// ---------------- launch ----------------
extern "C" void kernel_launch(void* const* d_in, const int* in_sizes, int n_in,
                              void* d_out, int out_size, void* d_ws, size_t ws_size,
                              hipStream_t stream) {
  const float* x      = (const float*)d_in[0];
  const float* memory = (const float*)d_in[1];
  const float* w1     = (const float*)d_in[2];
  const float* b1     = (const float*)d_in[3];
  const float* w2     = (const float*)d_in[4];
  const float* b2     = (const float*)d_in[5];
  float* out = (float*)d_out;

  char* ws = (char*)d_ws;
  unsigned short* memK = (unsigned short*)(ws + 0);          // 512 KB
  unsigned short* memT = (unsigned short*)(ws + 524288);     // 512 KB
  unsigned short* w1T  = (unsigned short*)(ws + 1048576);    // 1 MB
  unsigned short* w2T  = (unsigned short*)(ws + 2097152);    // 512 KB
  float*          xme  = (float*)(ws + 2621440);             // 64 KB
  unsigned short* gate = (unsigned short*)(ws + 3145728);    // 32 MB

  dmb_prep_kernel<<<4096, 256, 0, stream>>>(memory, w1, w2, memK, memT, w1T, w2T);
  dmb_xmean_kernel<<<64, 256, 0, stream>>>(x, xme);
  dmb_attn_kernel<<<(kB * kT) / 16, 256, 0, stream>>>(
      x, (const unsigned*)memK, (const unsigned*)memT, out);
  dmb_mlp_kernel<<<(kB * kT) / 16, 256, 0, stream>>>(
      out, (const unsigned*)w1T, (const unsigned*)w2T, b1, b2, gate);
  dmb_update_kernel<<<(kM * kC) / 256, 256, 0, stream>>>(
      memory, gate, xme, out + (size_t)kB * kT * 2 * kC);
}
